// CQAttention_16166256902618
// MI455X (gfx1250) — matrix operand order repaired and checked by reading, not compiled, not run on hardware
//
#include <hip/hip_runtime.h>
#include <hip/hip_bf16.h>
#include <cstdint>

// ---------------- problem constants (from reference) ----------------
#define B_   32
#define D_   128
#define CL_  2048
#define QL_  256

typedef __bf16 bf16_t;
typedef __attribute__((ext_vector_type(16))) __bf16 v16bf;
typedef __attribute__((ext_vector_type(8)))  __bf16 v8bf;
typedef __attribute__((ext_vector_type(8)))  float  v8f;

union AFrag { v16bf v; v8bf h[2]; };
union Acc   { v8f   v; float f[8]; };

__device__ __forceinline__ bf16_t to_bf16(float f) {
  unsigned u = __float_as_uint(f);
  u += 0x7FFFu + ((u >> 16) & 1u);          // round-to-nearest-even
  unsigned short hs = (unsigned short)(u >> 16);
  return __builtin_bit_cast(bf16_t, hs);
}

#define WMMA_BF16(Afr, Bfr, Cfr) \
  __builtin_amdgcn_wmma_f32_16x16x32_bf16(false, (Afr), false, (Bfr), (short)0, (Cfr), false, false)

// ---------------- workspace layout (bytes) ----------------
constexpr size_t SZ_QB     = (size_t)B_ * D_ * QL_ * 2;   // 2 MB  (bf16 q-sized)
constexpr size_t SZ_CB     = (size_t)B_ * D_ * CL_ * 2;   // 16 MB (bf16 c-sized)
constexpr size_t OFF_QBF   = 0;                            // bf16 q   [b][d][j]
constexpr size_t OFF_QBFT  = OFF_QBF   + SZ_QB;            // bf16 q^T [b][j][d]
constexpr size_t OFF_CBFWT = OFF_QBFT  + SZ_QB;            // bf16 (c*w_cq)^T [b][i][d]
constexpr size_t OFF_CBF   = OFF_CBFWT + SZ_CB;            // bf16 c   [b][d][i]
constexpr size_t OFF_QW    = OFF_CBF   + SZ_CB;            // f32 [b][j]  (incl bias)
constexpr size_t OFF_CW    = OFF_QW    + (size_t)B_*QL_*4; // f32 [b][i]
constexpr size_t OFF_RMAX  = OFF_CW    + (size_t)B_*CL_*4;
constexpr size_t OFF_RSUM  = OFF_RMAX  + (size_t)B_*CL_*4;
constexpr size_t OFF_CMAX  = OFF_RSUM  + (size_t)B_*CL_*4;
constexpr size_t OFF_CSUM  = OFF_CMAX  + (size_t)B_*QL_*4;
constexpr size_t OFF_S     = OFF_CSUM  + (size_t)B_*QL_*4; // f32 s [b][i][j]  (64 MB)
constexpr size_t OFF_E2    = OFF_S     + (size_t)B_*CL_*QL_*4; // bf16 exp(s-cmax)^T [b][j][i]
constexpr size_t OFF_S2TC  = OFF_E2    + (size_t)B_*QL_*CL_*2; // bf16 s2t_c^T [b][d][j]
constexpr size_t OFF_PMAX  = OFF_S2TC  + SZ_QB;            // f32 partial col max [8][B][QL]
constexpr size_t OFF_PSUM  = OFF_PMAX  + (size_t)8*B_*QL_*4;

// ============================================================
// K0q: per (b,j): qw = w_q . q(:,j) + bias ; bf16 copies of q (both layouts)
// ============================================================
__global__ __launch_bounds__(256) void k_prep_q(
    const float* __restrict__ q, const float* __restrict__ w, const float* __restrict__ bias,
    bf16_t* __restrict__ qbf, bf16_t* __restrict__ qbft, float* __restrict__ qw)
{
  const int idx = blockIdx.x * 256 + threadIdx.x;
  const int b = idx / QL_, j = idx % QL_;
  float acc = 0.f;
  #pragma unroll 4
  for (int d = 0; d < D_; ++d) {
    float v = q[((size_t)b * D_ + d) * QL_ + j];
    acc += v * w[d];
    bf16_t h = to_bf16(v);
    qbf [((size_t)b * D_  + d) * QL_ + j] = h;
    qbft[((size_t)b * QL_ + j) * D_  + d] = h;
  }
  qw[idx] = acc + bias[0];
}

// ============================================================
// K0c: per (b,i): cw = w_c . c(:,i) ; bf16 c ; bf16 (c*w_cq)^T ; out channel-0 = ct (exact fp32)
// ============================================================
__global__ __launch_bounds__(256) void k_prep_c(
    const float* __restrict__ c, const float* __restrict__ w,
    bf16_t* __restrict__ cbf, bf16_t* __restrict__ cbfwt, float* __restrict__ cw,
    float* __restrict__ out)
{
  const int idx = blockIdx.x * 256 + threadIdx.x;
  const int b = idx / CL_, i = idx % CL_;
  float acc = 0.f;
  #pragma unroll 4
  for (int d = 0; d < D_; ++d) {
    float v = c[((size_t)b * D_ + d) * CL_ + i];
    acc += v * w[D_ + d];
    cbf  [((size_t)b * D_  + d) * CL_ + i] = to_bf16(v);
    cbfwt[((size_t)b * CL_ + i) * D_  + d] = to_bf16(v * w[2 * D_ + d]);
    out[((size_t)b * 4 * D_ + d) * CL_ + i] = v;   // channel 0: ct copy
  }
  cw[idx] = acc;
}

// ============================================================
// K1: s = (c .* w_cq)^T @ q  + cw[i] + qw[j]   via bf16 WMMA (K=128)
//     One wave = 16 rows x 256 cols (16 tiles). Row softmax stats in-wave.
//     B fragments loaded in simultaneously-live groups of 4 so loads overlap WMMA.
// ============================================================
__global__ __launch_bounds__(256) void k_score(
    const bf16_t* __restrict__ cbfwt, const bf16_t* __restrict__ qbft,
    const float* __restrict__ cw, const float* __restrict__ qw,
    const int* __restrict__ c_mask,
    float* __restrict__ s, float* __restrict__ rowmax, float* __restrict__ rowsum)
{
  const int b    = blockIdx.x >> 4;          // 16 blocks/batch, 128 rows each
  const int it   = blockIdx.x & 15;
  const int wave = threadIdx.x >> 5;
  const int lane = threadIdx.x & 31;
  const int ln = lane & 15, hi = lane >> 4;
  const int ib = it * 128 + wave * 16;       // this wave's first row

  Acc acc[16];
  #pragma unroll
  for (int t = 0; t < 16; ++t)
    #pragma unroll
    for (int r = 0; r < 8; ++r) acc[t].f[r] = 0.f;

  const bf16_t* abase  = cbfwt + ((size_t)(b * CL_ + ib + ln)) * D_ + hi * 16;
  const bf16_t* bbase0 = qbft  + ((size_t)(b * QL_ + ln)) * D_ + hi * 16;

  AFrag a;
  a.h[0] = *(const v8bf*)(abase);
  a.h[1] = *(const v8bf*)(abase + 8);

  #pragma unroll 1
  for (int ks = 0; ks < 4; ++ks) {
    const int k0 = ks * 32;
    AFrag an = a;
    if (ks < 3) {                               // rotate A for next ks
      an.h[0] = *(const v8bf*)(abase + k0 + 32);
      an.h[1] = *(const v8bf*)(abase + k0 + 40);
    }
    #pragma unroll
    for (int jg = 0; jg < 4; ++jg) {
      const bf16_t* bp = bbase0 + k0 + (size_t)jg * 4 * (16 * D_);
      v16bf b0 = *(const v16bf*)(bp);
      v16bf b1 = *(const v16bf*)(bp + 1 * (16 * D_));
      v16bf b2 = *(const v16bf*)(bp + 2 * (16 * D_));
      v16bf b3 = *(const v16bf*)(bp + 3 * (16 * D_));
      acc[jg * 4 + 0].v = WMMA_BF16(a.v, b0, acc[jg * 4 + 0].v);
      acc[jg * 4 + 1].v = WMMA_BF16(a.v, b1, acc[jg * 4 + 1].v);
      acc[jg * 4 + 2].v = WMMA_BF16(a.v, b2, acc[jg * 4 + 2].v);
      acc[jg * 4 + 3].v = WMMA_BF16(a.v, b3, acc[jg * 4 + 3].v);
    }
    a = an;
  }

  // epilogue: rank-1 terms + store s + row stats
  float cwv[8];
  #pragma unroll
  for (int r = 0; r < 8; ++r) cwv[r] = cw[b * CL_ + ib + hi * 8 + r];
  float rmax[8], rsum[8];
  #pragma unroll
  for (int r = 0; r < 8; ++r) { rmax[r] = -3.0e38f; rsum[r] = 0.f; }

  #pragma unroll
  for (int jt = 0; jt < 16; ++jt) {
    const int j = jt * 16 + ln;
    const float qwv = qw[b * QL_ + j];
    #pragma unroll
    for (int r = 0; r < 8; ++r) {
      float v = acc[jt].f[r] + cwv[r] + qwv;
      acc[jt].f[r] = v;
      s[((size_t)b * CL_ + (ib + hi * 8 + r)) * QL_ + j] = v;
      rmax[r] = fmaxf(rmax[r], v);
    }
  }
  #pragma unroll
  for (int r = 0; r < 8; ++r)
    #pragma unroll
    for (int m = 1; m < 16; m <<= 1)
      rmax[r] = fmaxf(rmax[r], __shfl_xor(rmax[r], m, 32));
  #pragma unroll
  for (int jt = 0; jt < 16; ++jt)
    #pragma unroll
    for (int r = 0; r < 8; ++r)
      rsum[r] += __expf(acc[jt].f[r] - rmax[r]);
  #pragma unroll
  for (int r = 0; r < 8; ++r)
    #pragma unroll
    for (int m = 1; m < 16; m <<= 1)
      rsum[r] += __shfl_xor(rsum[r], m, 32);

  if (ln == 0) {
    #pragma unroll
    for (int r = 0; r < 8; ++r) {
      const int i = ib + hi * 8 + r;
      const int cm = c_mask[b * CL_ + i];
      rowmax[b * CL_ + i] = cm ? rmax[r] : 0.0f;
      rowsum[b * CL_ + i] = cm ? rsum[r] : (float)QL_;   // masked row -> uniform softmax
    }
  }
}

// ============================================================
// K2a: partial column softmax stats over 256-row chunks (8 chunks -> 256 WGs)
// ============================================================
__global__ __launch_bounds__(256) void k_colstats_part(
    const float* __restrict__ s, float* __restrict__ pmax, float* __restrict__ psum)
{
  const int b  = blockIdx.x >> 3;
  const int ch = blockIdx.x & 7;
  const int j  = threadIdx.x;
  const float* sp = s + (size_t)b * CL_ * QL_ + (size_t)ch * (CL_ / 8) * QL_ + j;
  float m = -3.0e38f, sum = 0.f;
  #pragma unroll 1
  for (int i = 0; i < CL_ / 8; i += 4) {      // 4 independent loads per rescale
    float v0 = sp[(size_t)(i + 0) * QL_];
    float v1 = sp[(size_t)(i + 1) * QL_];
    float v2 = sp[(size_t)(i + 2) * QL_];
    float v3 = sp[(size_t)(i + 3) * QL_];
    float lm = fmaxf(fmaxf(v0, v1), fmaxf(v2, v3));
    float nm = fmaxf(m, lm);
    sum = sum * __expf(m - nm)
        + __expf(v0 - nm) + __expf(v1 - nm) + __expf(v2 - nm) + __expf(v3 - nm);
    m = nm;
  }
  pmax[(ch * B_ + b) * QL_ + j] = m;
  psum[(ch * B_ + b) * QL_ + j] = sum;
}

// K2b: combine 8 partials, apply q_mask
__global__ __launch_bounds__(256) void k_colstats_combine(
    const float* __restrict__ pmax, const float* __restrict__ psum,
    const int* __restrict__ q_mask,
    float* __restrict__ cmax, float* __restrict__ csum)
{
  const int b = blockIdx.x;
  const int j = threadIdx.x;
  float m = -3.0e38f;
  float pm[8];
  #pragma unroll
  for (int ch = 0; ch < 8; ++ch) {
    pm[ch] = pmax[(ch * B_ + b) * QL_ + j];
    m = fmaxf(m, pm[ch]);
  }
  float sum = 0.f;
  #pragma unroll
  for (int ch = 0; ch < 8; ++ch)
    sum += psum[(ch * B_ + b) * QL_ + j] * __expf(pm[ch] - m);
  const int qm = q_mask[b * QL_ + j];
  cmax[b * QL_ + j] = qm ? m   : 0.0f;
  csum[b * QL_ + j] = qm ? sum : (float)CL_;              // masked col -> uniform softmax
}

// ============================================================
// K2.5: e2[b][j][i] = bf16( qm ? exp(s[i][j]-cmax[j]) : 1 )  (LDS transpose)
// ============================================================
__global__ __launch_bounds__(256) void k_e2(
    const float* __restrict__ s, const int* __restrict__ q_mask,
    const float* __restrict__ cmax, bf16_t* __restrict__ e2)
{
  __shared__ float tl[32 * 257];
  const int b  = blockIdx.x >> 6;
  const int i0 = (blockIdx.x & 63) * 32;
  const int tid = threadIdx.x;
  for (int f = tid; f < 32 * 256; f += 256) {
    int r = f >> 8, col = f & 255;
    tl[r * 257 + col] = s[((size_t)b * CL_ + i0 + r) * QL_ + col];
  }
  __syncthreads();
  for (int f = tid; f < 256 * 32; f += 256) {
    int j = f >> 5, ii = f & 31;
    float v = tl[ii * 257 + j];
    int qm = q_mask[b * QL_ + j];
    float e = qm ? __expf(v - cmax[b * QL_ + j]) : 1.0f;
    e2[((size_t)b * QL_ + j) * CL_ + i0 + ii] = to_bf16(e);
  }
}

// ============================================================
// K3: s2t_c^T[d][j] = ( c[d][:] @ e2[j][:]^T ) / csum[j]   (K = 2048, bf16 WMMA)
//     One group of 4 simultaneously-live B fragments per ks step.
// ============================================================
__global__ __launch_bounds__(256) void k_s2tc(
    const bf16_t* __restrict__ cbf, const bf16_t* __restrict__ e2,
    const float* __restrict__ csum, bf16_t* __restrict__ s2tc)
{
  const int b = blockIdx.x >> 2;
  const int g = blockIdx.x & 3;               // group of 4 j-tiles
  const int wave = threadIdx.x >> 5, lane = threadIdx.x & 31;
  const int ln = lane & 15, hi = lane >> 4;
  const int mt = wave;                        // d-tile

  Acc acc[4];
  #pragma unroll
  for (int t = 0; t < 4; ++t)
    #pragma unroll
    for (int r = 0; r < 8; ++r) acc[t].f[r] = 0.f;

  const bf16_t* abase  = cbf + ((size_t)b * D_ + mt * 16 + ln) * CL_ + hi * 16;
  const bf16_t* bbase0 = e2  + ((size_t)b * QL_ + g * 64 + ln) * CL_ + hi * 16;

  AFrag a;
  a.h[0] = *(const v8bf*)(abase);
  a.h[1] = *(const v8bf*)(abase + 8);

  #pragma unroll 1
  for (int ks = 0; ks < 64; ++ks) {
    const int k0 = ks * 32;
    AFrag an = a;
    if (ks < 63) {
      an.h[0] = *(const v8bf*)(abase + k0 + 32);
      an.h[1] = *(const v8bf*)(abase + k0 + 40);
    }
    const bf16_t* bp = bbase0 + k0;
    v16bf b0 = *(const v16bf*)(bp);
    v16bf b1 = *(const v16bf*)(bp + 1 * (16 * CL_));
    v16bf b2 = *(const v16bf*)(bp + 2 * (16 * CL_));
    v16bf b3 = *(const v16bf*)(bp + 3 * (16 * CL_));
    acc[0].v = WMMA_BF16(a.v, b0, acc[0].v);
    acc[1].v = WMMA_BF16(a.v, b1, acc[1].v);
    acc[2].v = WMMA_BF16(a.v, b2, acc[2].v);
    acc[3].v = WMMA_BF16(a.v, b3, acc[3].v);
    a = an;
  }
  #pragma unroll
  for (int t = 0; t < 4; ++t) {
    const int j = (g * 4 + t) * 16 + ln;
    const float inv = 1.0f / csum[b * QL_ + j];
    #pragma unroll
    for (int r = 0; r < 8; ++r) {
      const int d = mt * 16 + hi * 8 + r;
      s2tc[((size_t)b * D_ + d) * QL_ + j] = to_bf16(acc[t].f[r] * inv);
    }
  }
}

// ============================================================
// K4: s1 tile (64 rows) -> LDS bf16 ; A = s1 @ q^T ; Bt = s1 @ s2t_c ;
//     group of 8 live B fragments per ks ; out channels 1..3 coalesced via LDS transpose.
// ============================================================
__global__ __launch_bounds__(256) void k_out(
    const float* __restrict__ s, const float* __restrict__ c, const int* __restrict__ c_mask,
    const float* __restrict__ rowmax, const float* __restrict__ rowsum,
    const bf16_t* __restrict__ qbf, const bf16_t* __restrict__ s2tc,
    float* __restrict__ out)
{
  __shared__ __align__(16) unsigned char smem[64 * 264 * 2];   // 33,792 B, dual-purpose
  bf16_t* e1  = (bf16_t*)smem;   // [64][264] bf16 s1 tile
  float* tbuf = (float*)smem;    // [128][66] f32 transpose buffer (same bytes)

  const int b  = blockIdx.x >> 5;
  const int i0 = (blockIdx.x & 31) * 64;
  const int tid = threadIdx.x;
  const int wave = tid >> 5, lane = tid & 31, ln = lane & 15, hi = lane >> 4;

  // phase 1: s1 = exp(s - rowmax) (c_mask gated) -> LDS bf16
  for (int f = tid; f < 64 * 256; f += 256) {
    int r = f >> 8, col = f & 255;
    int i = i0 + r;
    float v = s[((size_t)b * CL_ + i) * QL_ + col];
    int cm = c_mask[b * CL_ + i];
    float e = cm ? __expf(v - rowmax[b * CL_ + i]) : 1.0f;
    e1[r * 264 + col] = to_bf16(e);
  }
  __syncthreads();

  // phase 2: both WMMA products share the A fragment (K = 256 over j)
  const int rb = wave & 3, half = wave >> 2;
  const int rloc = rb * 16;
  Acc accA[4], accB[4];
  #pragma unroll
  for (int t = 0; t < 4; ++t)
    #pragma unroll
    for (int r = 0; r < 8; ++r) { accA[t].f[r] = 0.f; accB[t].f[r] = 0.f; }

  const bf16_t* qbase0 = qbf  + ((size_t)b * D_ + half * 64 + ln) * QL_ + hi * 16;
  const bf16_t* sbase0 = s2tc + ((size_t)b * D_ + half * 64 + ln) * QL_ + hi * 16;

  #pragma unroll 1
  for (int ks = 0; ks < 8; ++ks) {
    const int k0 = ks * 32;
    const bf16_t* ar = e1 + (rloc + ln) * 264 + (k0 + hi * 16);
    AFrag a;
    a.h[0] = *(const v8bf*)(ar);
    a.h[1] = *(const v8bf*)(ar + 8);
    const bf16_t* qp = qbase0 + k0;
    const bf16_t* sp = sbase0 + k0;
    v16bf q0 = *(const v16bf*)(qp);
    v16bf q1 = *(const v16bf*)(qp + 1 * (16 * QL_));
    v16bf q2 = *(const v16bf*)(qp + 2 * (16 * QL_));
    v16bf q3 = *(const v16bf*)(qp + 3 * (16 * QL_));
    v16bf s0 = *(const v16bf*)(sp);
    v16bf s1 = *(const v16bf*)(sp + 1 * (16 * QL_));
    v16bf s2 = *(const v16bf*)(sp + 2 * (16 * QL_));
    v16bf s3 = *(const v16bf*)(sp + 3 * (16 * QL_));
    accA[0].v = WMMA_BF16(a.v, q0, accA[0].v);
    accA[1].v = WMMA_BF16(a.v, q1, accA[1].v);
    accA[2].v = WMMA_BF16(a.v, q2, accA[2].v);
    accA[3].v = WMMA_BF16(a.v, q3, accA[3].v);
    accB[0].v = WMMA_BF16(a.v, s0, accB[0].v);
    accB[1].v = WMMA_BF16(a.v, s1, accB[1].v);
    accB[2].v = WMMA_BF16(a.v, s2, accB[2].v);
    accB[3].v = WMMA_BF16(a.v, s3, accB[3].v);
  }

  float rinv[8];
  #pragma unroll
  for (int r = 0; r < 8; ++r)
    rinv[r] = 1.0f / rowsum[b * CL_ + i0 + rloc + hi * 8 + r];

  // A output: tbuf transpose -> coalesced stores of A and ct*A
  __syncthreads();
  #pragma unroll
  for (int t = 0; t < 4; ++t) {
    const int dd = (half * 4 + t) * 16 + ln;
    #pragma unroll
    for (int r = 0; r < 8; ++r)
      tbuf[dd * 66 + (rloc + hi * 8 + r)] = accA[t].f[r] * rinv[r];
  }
  __syncthreads();
  for (int f = tid; f < 128 * 64; f += 256) {
    int dd = f >> 6, ii = f & 63;
    float a  = tbuf[dd * 66 + ii];
    float cv = c[((size_t)b * D_ + dd) * CL_ + i0 + ii];
    size_t o = ((size_t)b * 4 * D_) * CL_ + i0 + ii;
    out[o + (size_t)(    D_ + dd) * CL_] = a;
    out[o + (size_t)(2 * D_ + dd) * CL_] = cv * a;
  }
  __syncthreads();

  // B output: ct * B_attn
  #pragma unroll
  for (int t = 0; t < 4; ++t) {
    const int dd = (half * 4 + t) * 16 + ln;
    #pragma unroll
    for (int r = 0; r < 8; ++r)
      tbuf[dd * 66 + (rloc + hi * 8 + r)] = accB[t].f[r] * rinv[r];
  }
  __syncthreads();
  for (int f = tid; f < 128 * 64; f += 256) {
    int dd = f >> 6, ii = f & 63;
    float bv = tbuf[dd * 66 + ii];
    float cv = c[((size_t)b * D_ + dd) * CL_ + i0 + ii];
    out[((size_t)b * 4 * D_ + 3 * D_ + dd) * CL_ + i0 + ii] = cv * bv;
  }
}

// ============================================================
extern "C" void kernel_launch(void* const* d_in, const int* in_sizes, int n_in,
                              void* d_out, int out_size, void* d_ws, size_t ws_size,
                              hipStream_t stream)
{
  const float* c      = (const float*)d_in[0];
  const int*   c_mask = (const int*)  d_in[1];
  const float* q      = (const float*)d_in[2];
  const int*   q_mask = (const int*)  d_in[3];
  const float* w      = (const float*)d_in[4];
  const float* bias   = (const float*)d_in[5];
  float* out = (float*)d_out;

  char* ws = (char*)d_ws;
  bf16_t* qbf   = (bf16_t*)(ws + OFF_QBF);
  bf16_t* qbft  = (bf16_t*)(ws + OFF_QBFT);
  bf16_t* cbfwt = (bf16_t*)(ws + OFF_CBFWT);
  bf16_t* cbf   = (bf16_t*)(ws + OFF_CBF);
  float*  qw    = (float*) (ws + OFF_QW);
  float*  cw    = (float*) (ws + OFF_CW);
  float*  rmax  = (float*) (ws + OFF_RMAX);
  float*  rsum  = (float*) (ws + OFF_RSUM);
  float*  cmax  = (float*) (ws + OFF_CMAX);
  float*  csum  = (float*) (ws + OFF_CSUM);
  float*  sbuf  = (float*) (ws + OFF_S);
  bf16_t* e2    = (bf16_t*)(ws + OFF_E2);
  bf16_t* s2tc  = (bf16_t*)(ws + OFF_S2TC);
  float*  pmax  = (float*) (ws + OFF_PMAX);
  float*  psum  = (float*) (ws + OFF_PSUM);

  k_prep_q          <<<dim3((B_ * QL_) / 256), 256, 0, stream>>>(q, w, bias, qbf, qbft, qw);
  k_prep_c          <<<dim3((B_ * CL_) / 256), 256, 0, stream>>>(c, w, cbf, cbfwt, cw, out);
  k_score           <<<dim3(B_ * 16),          256, 0, stream>>>(cbfwt, qbft, cw, qw, c_mask, sbuf, rmax, rsum);
  k_colstats_part   <<<dim3(B_ * 8),           256, 0, stream>>>(sbuf, pmax, psum);
  k_colstats_combine<<<dim3(B_),               256, 0, stream>>>(pmax, psum, q_mask, cmax, csum);
  k_e2              <<<dim3(B_ * 64),          256, 0, stream>>>(sbuf, q_mask, cmax, e2);
  k_s2tc            <<<dim3(B_ * 4),           256, 0, stream>>>(cbf, e2, csum, s2tc);
  k_out             <<<dim3(B_ * 32),          256, 0, stream>>>(sbuf, c, c_mask, rmax, rsum, qbf, s2tc, out);
}
